// LSTMStack_49228915147125
// MI455X (gfx1250) — matrix operand order repaired and checked by reading, not compiled, not run on hardware
//
#include <hip/hip_runtime.h>
#include <cstdint>

typedef __attribute__((ext_vector_type(8)))  float  v8f;
typedef __attribute__((ext_vector_type(8)))  __bf16 v8bf;
typedef __attribute__((ext_vector_type(16))) __bf16 v16bf;
typedef int v4i_b128 __attribute__((vector_size(16)));   // pointee type of async-LDS builtins

#define B_ 32
#define T_ 512
#define D_ 1024
#define H_ 1024
#define L_ 3
#define BH_ (B_*H_)

#define LDSV(p) ((__attribute__((address_space(3))) v4i_b128*)(p))
#define GLBV(p) ((__attribute__((address_space(1))) v4i_b128*)(p))

__device__ __forceinline__ float sigm_f(float x) { return 1.0f / (1.0f + __expf(-x)); }

// ---------------------------------------------------------------------------
// Pack [Wi;Wh] (fp32, K=2048 x N=4096 per layer) into bf16 WMMA B-fragments.
// Fragment (l, nt, kt) = 32 lanes x 16 bf16 (1KB); lane ln supplies column
// N = nt*16 + (ln&15); element e=2j+h holds K = kt*32 + (ln>>4)*16 + e.
// (Contiguous k order, not the ISA table order; lstm_step's A loads match it.)
// ---------------------------------------------------------------------------
__global__ void pack_weights(const float* __restrict__ Wi,
                             const float* __restrict__ Wh,
                             __bf16* __restrict__ Wp) {
  long id = (long)blockIdx.x * blockDim.x + threadIdx.x;
  const long TOT = (long)L_ * 256 * 64 * 32 * 8;   // dwords (2 bf16 each)
  if (id >= TOT) return;
  int j  = (int)(id & 7);  long r = id >> 3;
  int ln = (int)(r & 31);  r >>= 5;
  int kt = (int)(r & 63);  r >>= 6;
  int nt = (int)(r & 255); r >>= 8;
  int l  = (int)r;
  int n     = nt * 16 + (ln & 15);
  int kbase = kt * 32 + (ln >> 4) * 16 + 2 * j;
  long dst  = ((((long)l * 256 + nt) * 64 + kt) << 9) + (long)ln * 16 + j * 2;
#pragma unroll
  for (int h = 0; h < 2; ++h) {
    int k = kbase + h;
    float v = (k < D_) ? Wi[((long)l * D_ + k) * (4 * H_) + n]
                       : Wh[((long)l * H_ + (k - D_)) * (4 * H_) + n];
    Wp[dst + h] = (__bf16)v;
  }
}

__global__ void cvt_x(const float* __restrict__ x, __bf16* __restrict__ xbf) {
  long id = (long)blockIdx.x * blockDim.x + threadIdx.x;
  if (id < (long)B_ * T_ * D_) xbf[id] = (__bf16)x[id];
}

__global__ void zero_state(__bf16* __restrict__ hbuf, float* __restrict__ c) {
  int id = blockIdx.x * blockDim.x + threadIdx.x;
  if (id < L_ * 2 * BH_) hbuf[id] = (__bf16)0.0f;
  if (id < L_ * BH_)     c[id] = 0.0f;
}

// ---------------------------------------------------------------------------
// One (timestep, layer) LSTM cell step.
// Block = 8 waves, one 16-row m-tile; A panel (16 x 2048 bf16 = 64KB) staged
// in LDS via async global->LDS copies with an XOR-by-row 16B-chunk swizzle
// (conflict-free ds_load_b128). B fragments stream from the L2-resident
// packed weights with software double-buffering so WMMA overlaps the loads.
// ---------------------------------------------------------------------------
__global__ __launch_bounds__(256)
void lstm_step(const __bf16* __restrict__ A0, long sA0,
               const __bf16* __restrict__ A1, long sA1,
               const __bf16* __restrict__ Wl,
               const float*  __restrict__ bias,
               float*  __restrict__ c,
               __bf16* __restrict__ h_bf,
               float*  __restrict__ h_f32,
               float*  __restrict__ y,
               int t, int is_top) {
  __shared__ __bf16 sA[16 * 2048];                   // exactly 64 KB

  int tid  = threadIdx.x;
  int lane = tid & 31;
  int wid  = (blockIdx.x << 3) + (tid >> 5);         // 0..127
  int m0   = (wid >> 6) << 4;                        // 0 or 16 (uniform/block)
  int nth  = wid & 63;                               // 16-col tile within H
  int lh   = lane >> 4;                              // lane half
  int r16  = lane & 15;

  // ---- Stage A panel: thread -> 16 chunks of 16B, chunk c of row r lands at
  // ---- LDS element offset r*2048 + ((c ^ r) * 8).
  {
    int r   = tid >> 4;                              // 0..15  (row in panel)
    int tir = tid & 15;                              // 0..15  (16 chunks each)
#pragma unroll
    for (int i = 0; i < 16; ++i) {
      int cch = tir * 16 + i;                        // chunk 0..255
      int k   = cch * 8;                             // col 0..2047
      const __bf16* gsrc = (k < 1024) ? (A0 + (long)(m0 + r) * sA0 + k)
                                      : (A1 + (long)(m0 + r) * sA1 + (k - 1024));
      __bf16* ldst = sA + r * 2048 + ((cch ^ r) * 8);
#if __has_builtin(__builtin_amdgcn_global_load_async_to_lds_b128)
      __builtin_amdgcn_global_load_async_to_lds_b128(GLBV(gsrc), LDSV(ldst), 0, 0);
#else
      *(v8bf*)ldst = *(const v8bf*)gsrc;
#endif
    }
#if __has_builtin(__builtin_amdgcn_global_load_async_to_lds_b128)
#if __has_builtin(__builtin_amdgcn_s_wait_asynccnt)
    __builtin_amdgcn_s_wait_asynccnt(0);
#else
    asm volatile("s_wait_asynccnt 0" ::: "memory");
#endif
#endif
    __syncthreads();
  }

  v8f acc[4];
#pragma unroll
  for (int g = 0; g < 4; ++g) acc[g] = (v8f){0, 0, 0, 0, 0, 0, 0, 0};

  const __bf16* srow = sA + r16 * 2048;              // this lane's A row

  auto loadA = [&](int kt) -> v16bf {
    int c0 = kt * 4 + lh * 2;                        // chunk of K kb+lh*16
    int c1 = c0 + 1;                                 // chunk of K kb+lh*16+8
    v8bf lo = *(const v8bf*)(srow + ((c0 ^ r16) * 8));
    v8bf hi = *(const v8bf*)(srow + ((c1 ^ r16) * 8));
    return __builtin_shufflevector(lo, hi, 0,1,2,3,4,5,6,7,8,9,10,11,12,13,14,15);
  };
  auto loadB = [&](int g, int kt) -> v16bf {
    long frag = (((long)(g * 64 + nth) * 64 + kt) << 9) + (long)lane * 16;
    return *(const v16bf*)(Wl + frag);
  };

  // ---- Software-pipelined K loop: fetch kt+1 while WMMAing kt.
  v16bf aC = loadA(0);
  v16bf b0 = loadB(0, 0), b1 = loadB(1, 0), b2 = loadB(2, 0), b3 = loadB(3, 0);
#pragma unroll 4
  for (int kt = 0; kt < 64; ++kt) {
    v16bf aN = aC, n0 = b0, n1 = b1, n2 = b2, n3 = b3;
    if (kt + 1 < 64) {
      aN = loadA(kt + 1);
      n0 = loadB(0, kt + 1); n1 = loadB(1, kt + 1);
      n2 = loadB(2, kt + 1); n3 = loadB(3, kt + 1);
    }
    acc[0] = __builtin_amdgcn_wmma_f32_16x16x32_bf16(false, aC, false, b0, (short)0, acc[0], false, false);
    acc[1] = __builtin_amdgcn_wmma_f32_16x16x32_bf16(false, aC, false, b1, (short)0, acc[1], false, false);
    acc[2] = __builtin_amdgcn_wmma_f32_16x16x32_bf16(false, aC, false, b2, (short)0, acc[2], false, false);
    acc[3] = __builtin_amdgcn_wmma_f32_16x16x32_bf16(false, aC, false, b3, (short)0, acc[3], false, false);
    aC = aN; b0 = n0; b1 = n1; b2 = n2; b3 = n3;
  }

  // ---- Epilogue: gate math on the 16x16 f32 C layout.
  int nh = nth * 16 + r16;                           // column within H
  float bi = bias[0 * H_ + nh];
  float bf = bias[1 * H_ + nh];
  float bg = bias[2 * H_ + nh];
  float bo = bias[3 * H_ + nh];
  int rbase = m0 + lh * 8;                           // batch row of acc[*][0]
#pragma unroll
  for (int r = 0; r < 8; ++r) {
    int bidx = rbase + r;
    float iv = acc[0][r] + bi;
    float fv = acc[1][r] + bf;
    float gv = acc[2][r] + bg;
    float ov = acc[3][r] + bo;
    long off = (long)bidx * H_ + nh;
    float cold = c[off];
    float cn = sigm_f(fv) * cold + sigm_f(iv) * tanhf(gv);
    float hn = sigm_f(ov) * tanhf(cn);
    c[off]     = cn;
    h_bf[off]  = (__bf16)hn;
    h_f32[off] = hn;
    if (is_top) y[((long)bidx * T_ + t) * H_ + nh] = hn;
  }
}

// ---------------------------------------------------------------------------
extern "C" void kernel_launch(void* const* d_in, const int* in_sizes, int n_in,
                              void* d_out, int out_size, void* d_ws, size_t ws_size,
                              hipStream_t stream) {
  const float* x  = (const float*)d_in[0];   // (B,T,D)
  const float* Wi = (const float*)d_in[1];   // (L,D,4H)
  const float* Wh = (const float*)d_in[2];   // (L,H,4H)
  const float* b  = (const float*)d_in[3];   // (L,4H)

  char* ws = (char*)d_ws;
  const long WP_BYTES = (long)L_ * 2048 * 4096 * 2;          // 50,331,648
  const long XB_BYTES = (long)B_ * T_ * D_ * 2;              // 33,554,432
  __bf16* Wp   = (__bf16*)ws;
  __bf16* xbf  = (__bf16*)(ws + WP_BYTES);
  __bf16* hbuf = (__bf16*)(ws + WP_BYTES + XB_BYTES);        // [l][pp][B*H]

  float* y  = (float*)d_out;                                 // (B,T,H)
  float* hT = y + (long)B_ * T_ * H_;                        // (L,B,H)
  float* cT = hT + (long)L_ * BH_;                           // (L,B,H) = live c

  {
    const long TOT = (long)L_ * 256 * 64 * 32 * 8;
    pack_weights<<<(int)((TOT + 255) / 256), 256, 0, stream>>>(Wi, Wh, Wp);
  }
  {
    const long N = (long)B_ * T_ * D_;
    cvt_x<<<(int)((N + 255) / 256), 256, 0, stream>>>(x, xbf);
  }
  {
    const int N = L_ * 2 * BH_;
    zero_state<<<(N + 255) / 256, 256, 0, stream>>>(hbuf, cT);
  }

  for (int t = 0; t < T_; ++t) {
    int p = t & 1;
    for (int l = 0; l < L_; ++l) {
      const __bf16* A0; long sA0;
      if (l == 0) { A0 = xbf + (long)t * D_;                         sA0 = (long)T_ * D_; }
      else        { A0 = hbuf + (long)((l - 1) * 2 + (1 - p)) * BH_; sA0 = H_; }
      const __bf16* A1 = hbuf + (long)(l * 2 + p) * BH_;
      const __bf16* Wl = Wp + (long)l * 2048 * 4096;
      const float*  bl = b + (long)l * 4 * H_;
      float*  cl = cT + (long)l * BH_;
      __bf16* hb = hbuf + (long)(l * 2 + (1 - p)) * BH_;
      float*  hf = hT + (long)l * BH_;
      lstm_step<<<16, 256, 0, stream>>>(A0, sA0, A1, (long)H_, Wl, bl,
                                        cl, hb, hf, y, t, (l == L_ - 1) ? 1 : 0);
    }
  }
}
